// ScalableCritic_2388001817196
// MI455X (gfx1250) — compile-verified, matrix operand order checked by reading, not run on hardware
//
#include <hip/hip_runtime.h>
#include <hip/hip_bf16.h>

typedef _Float16 v16h __attribute__((ext_vector_type(16)));
typedef _Float16 v8h  __attribute__((ext_vector_type(8)));
typedef float    v8f  __attribute__((ext_vector_type(8)));
typedef unsigned u32x4 __attribute__((ext_vector_type(4)));
typedef int      i32x8 __attribute__((ext_vector_type(8)));
typedef int      i32x4 __attribute__((ext_vector_type(4)));

#define LN_EPS   1e-5f
#define NEG_FILL -1e9f

#if defined(__has_builtin)
#if __has_builtin(__builtin_amdgcn_tensor_load_to_lds) && __has_builtin(__builtin_amdgcn_s_wait_tensorcnt)
#define HAVE_TDM 1
#endif
#endif

__device__ __forceinline__ unsigned fkey(float f) {
  unsigned u = __float_as_uint(f);
  return (u & 0x80000000u) ? ~u : (u | 0x80000000u);
}
__device__ __forceinline__ float fdec(unsigned k) {
  unsigned u = (k & 0x80000000u) ? (k & 0x7fffffffu) : ~k;
  return __uint_as_float(u);
}
__device__ __forceinline__ float lrelu(float x) { return x >= 0.f ? x : 0.2f * x; }
__device__ __forceinline__ float hsum16(float v) {  // sum within each 16-lane half
  v += __shfl_xor(v, 1);
  v += __shfl_xor(v, 2);
  v += __shfl_xor(v, 4);
  v += __shfl_xor(v, 8);
  return v;
}

// ---------------------------------------------------------------------------
// Kernel 0: init gf keys to key(-1e9), zero |score| accumulator
// ---------------------------------------------------------------------------
__global__ void k_init(unsigned* __restrict__ gfkey, float* __restrict__ sumAbs, int total) {
  int i = blockIdx.x * blockDim.x + threadIdx.x;
  if (i < total) gfkey[i] = fkey(NEG_FILL);
  if (i == 0) *sumAbs = 0.f;
}

// ---------------------------------------------------------------------------
// Kernel 1: fused point-net (7->64 LN lrelu ->128 LN lrelu) + segmented max
// Block = 256 threads (8 waves), 128 muon rows per block.
// ---------------------------------------------------------------------------
__global__ __launch_bounds__(256) void k_pointnet(
    const float* __restrict__ muons, const int* __restrict__ bindex,
    const float* __restrict__ cond,
    const float* __restrict__ w1, const float* __restrict__ b1,
    const float* __restrict__ g1, const float* __restrict__ be1,
    const float* __restrict__ w2, const float* __restrict__ b2,
    const float* __restrict__ g2, const float* __restrict__ be2,
    unsigned* __restrict__ gfkey, int N)
{
  __shared__ __align__(16) float xmu[128 * 3];      // TDM staging of muon tile
  __shared__ int   bidx[128];
  __shared__ float xin[128][8];
  __shared__ float w1s[7 * 64];
  __shared__ float b1s[64], g1s[64], be1s[64];
  __shared__ float b2s[128], g2s[128], be2s[128];
  __shared__ float hbuf[128 * 129];                 // h1 view: r*65+j ; h2 view: r*129+c
  __shared__ __align__(16) _Float16 hA[128 * 72];   // layer1 activations, f16, WMMA-A friendly
  __shared__ __align__(32) _Float16 Bp[2][8][32][16]; // w2 packed in WMMA-B per-lane layout

  const int t    = threadIdx.x;
  const int lane = t & 31;
  const int wv   = t >> 5;
  const int rowbase = blockIdx.x * 128;

#ifdef HAVE_TDM
  // ---- issue async TDM load of this block's muon tile (overlaps weight staging)
  if (t < 32) {
    unsigned lds_off = (unsigned)__builtin_amdgcn_readfirstlane(
        (int)(unsigned)(unsigned long long)(void*)xmu);
    unsigned long long ga = (unsigned long long)(const void*)(muons + (size_t)rowbase * 3);
    unsigned ga_lo = (unsigned)__builtin_amdgcn_readfirstlane((int)(unsigned)ga);
    unsigned ga_hi = (unsigned)__builtin_amdgcn_readfirstlane((int)(unsigned)(ga >> 32));
    // D# group0: count=1 | lds_addr | global_addr[56:0] | type=2
    u32x4 g0 = { 1u, lds_off, ga_lo, (ga_hi & 0x01FFFFFFu) | (2u << 30) };
    // D# group1: data_size=4B(code 2); tensor_dim0 = tile_dim0 = 384 elements (1-D tile)
    i32x8 g1 = { (int)(2u << 16),
                 (int)(384u << 16),   // tensor_dim0[15:0] at bits 63:48
                 0,                   // tensor_dim0[31:16]=0, tensor_dim1[15:0]=0
                 (int)(384u << 16),   // tile_dim0 at bits 127:112
                 0, 0, 0, 0 };
    i32x4 z4 = { 0, 0, 0, 0 };
    i32x8 z8 = { 0, 0, 0, 0, 0, 0, 0, 0 };
    __builtin_amdgcn_tensor_load_to_lds(g0, g1, z4, z4, z8, 0);
  }
#endif

  // ---- phase 0: stage weights ----
  for (int i = t; i < 7 * 64; i += 256) w1s[i] = w1[i];
  for (int i = t; i < 64; i += 256) { b1s[i] = b1[i]; g1s[i] = g1[i]; be1s[i] = be1[i]; }
  for (int i = t; i < 128; i += 256) { b2s[i] = b2[i]; g2s[i] = g2[i]; be2s[i] = be2[i]; }
  for (int e = t; e < 2 * 8 * 32 * 16; e += 256) {
    int i  = e & 15;
    int ln = (e >> 4) & 31;
    int ct = (e >> 9) & 7;
    int ks = (e >> 12) & 1;
    int k  = ks * 32 + ((ln >> 4) << 3) + (i < 8 ? i : 16 + (i - 8));
    int c  = ct * 16 + (ln & 15);
    Bp[ks][ct][ln][i] = (_Float16)w2[k * 128 + c];
  }
  if (t < 128) {
    int gr = rowbase + t; if (gr >= N) gr = N - 1;
    bidx[t] = bindex[gr];
  }
  // prefetch next tile's muon features into cache hierarchy
  if (rowbase + 128 < N)
    __builtin_prefetch(&muons[(size_t)(rowbase + 128) * 3 + (t & 255)], 0, 0);

#ifdef HAVE_TDM
  if (t < 32) __builtin_amdgcn_s_wait_tensorcnt(0);
#else
  for (int e = t; e < 384; e += 256) {
    size_t gi = (size_t)rowbase * 3 + e;
    if (gi >= (size_t)N * 3) gi = (size_t)N * 3 - 1;
    xmu[e] = muons[gi];
  }
#endif
  __syncthreads();

  // ---- phase 0b: inputs [128][7] = muons(3) ++ gathered conditions(4) ----
  for (int e = t; e < 128 * 8; e += 256) {
    int r = e >> 3, i = e & 7;
    float v = 0.f;
    if (i < 3)      v = xmu[r * 3 + i];
    else if (i < 7) v = cond[(size_t)bidx[r] * 4 + (i - 3)];
    xin[r][i] = v;
  }
  __syncthreads();

  // ---- phase 1: layer1 (7 -> 64), VALU f32 ----
  {
    int r  = t >> 1;
    int c0 = (t & 1) * 32;
    float xi[7];
    #pragma unroll
    for (int i = 0; i < 7; ++i) xi[i] = xin[r][i];
    for (int j = c0; j < c0 + 32; ++j) {
      float a = b1s[j];
      #pragma unroll
      for (int i = 0; i < 7; ++i) a += xi[i] * w1s[i * 64 + j];
      hbuf[r * 65 + j] = a;
    }
  }
  __syncthreads();

  // ---- phase 2: LN(64) + lrelu -> hA (f16); 2 threads per row ----
  {
    int r  = t >> 1;
    int j0 = (t & 1) * 32;
    float s = 0.f, sq = 0.f;
    for (int j = j0; j < j0 + 32; ++j) {
      float x = hbuf[r * 65 + j];
      s += x; sq += x * x;
    }
    s  += __shfl_xor(s, 1);     // partner thread handles the other 32 columns
    sq += __shfl_xor(sq, 1);
    float m  = s * (1.f / 64.f);
    float vv = sq * (1.f / 64.f) - m * m;
    float inv = rsqrtf(vv + LN_EPS);
    for (int j = j0; j < j0 + 32; ++j) {
      float y = (hbuf[r * 65 + j] - m) * inv * g1s[j] + be1s[j];
      hA[r * 72 + j] = (_Float16)lrelu(y);
    }
  }
  __syncthreads();

  // ---- phase 3: WMMA layer2 (64 -> 128) + bias; phase 4: LN(128)+lrelu ----
  {
    const int rw  = wv * 16 + (lane & 15);
    const int kb8 = (lane >> 4) << 3;   // 0 for lanes 0-15, 8 for lanes 16-31
    v16h A[2];
    #pragma unroll
    for (int ks = 0; ks < 2; ++ks) {
      int kb = ks * 32 + kb8;
      union { v16h v; v8h h[2]; } u;
      u.h[0] = *(const v8h*)&hA[rw * 72 + kb];
      u.h[1] = *(const v8h*)&hA[rw * 72 + kb + 16];
      A[ks] = u.v;
    }
    v8f acc[8];
    #pragma unroll
    for (int ct = 0; ct < 8; ++ct) {
      float bz = b2s[ct * 16 + (lane & 15)];
      v8f a = {bz, bz, bz, bz, bz, bz, bz, bz};
      acc[ct] = a;
    }
    #pragma unroll
    for (int ct = 0; ct < 8; ++ct) {
      #pragma unroll
      for (int ks = 0; ks < 2; ++ks) {
        union { v16h v; v8h h[2]; } u;
        u.h[0] = *(const v8h*)&Bp[ks][ct][lane][0];
        u.h[1] = *(const v8h*)&Bp[ks][ct][lane][8];
        acc[ct] = __builtin_amdgcn_wmma_f32_16x16x32_f16(
            false, A[ks], false, u.v, (short)0, acc[ct], false, false);
      }
    }
    // LN over 128 cols directly on C layout: row = wv*16 + v + kb8, col = ct*16 + lane&15
    float m[8], inv[8];
    #pragma unroll
    for (int v = 0; v < 8; ++v) {
      float s = 0.f, sq = 0.f;
      #pragma unroll
      for (int ct = 0; ct < 8; ++ct) { float x = acc[ct][v]; s += x; sq += x * x; }
      s = hsum16(s); sq = hsum16(sq);
      float mm = s * (1.f / 128.f);
      float vv = sq * (1.f / 128.f) - mm * mm;
      m[v] = mm; inv[v] = rsqrtf(vv + LN_EPS);
    }
    #pragma unroll
    for (int ct = 0; ct < 8; ++ct) {
      int c = ct * 16 + (lane & 15);
      float g = g2s[c], be = be2s[c];
      #pragma unroll
      for (int v = 0; v < 8; ++v) {
        int r = wv * 16 + v + kb8;
        float x = (acc[ct][v] - m[v]) * inv[v] * g + be;
        x = lrelu(x);
        if (!__builtin_isfinite(x)) x = 0.f;       // nan_to_num(0,0,0)
        hbuf[r * 129 + c] = x;                     // h2 view (h1 region is dead)
      }
    }
  }
  __syncthreads();

  // ---- phase 5: segmented max over sorted batch_index (few atomics/block) ----
  {
    int c  = t & 127;
    int r0 = (t >> 7) * 64;
    int curEv = bidx[r0];
    float curMax = NEG_FILL;
    for (int r = r0; r < r0 + 64; ++r) {
      int ev = bidx[r];
      if (ev != curEv) {
        atomicMax(&gfkey[(size_t)curEv * 128 + c], fkey(curMax));
        curEv = ev; curMax = NEG_FILL;
      }
      curMax = fmaxf(curMax, hbuf[r * 129 + c]);
    }
    atomicMax(&gfkey[(size_t)curEv * 128 + c], fkey(curMax));
  }
}

// ---------------------------------------------------------------------------
// Kernel 2: decision net (132->128 LN lrelu ->64 LN lrelu ->1), WMMA
// Block = 256 threads, 128 event rows per block.
// ---------------------------------------------------------------------------
__global__ __launch_bounds__(256) void k_decision(
    const unsigned* __restrict__ gfkey, const float* __restrict__ cond,
    const float* __restrict__ w3, const float* __restrict__ b3,
    const float* __restrict__ g3, const float* __restrict__ be3,
    const float* __restrict__ w4, const float* __restrict__ b4,
    const float* __restrict__ g4, const float* __restrict__ be4,
    const float* __restrict__ w5, const float* __restrict__ b5,
    float* __restrict__ score, float* __restrict__ sumAbs, int B)
{
  __shared__ __align__(16) _Float16 gfA[128 * 136];   // also reused as h4 (f32) later
  __shared__ __align__(32) _Float16 W3p[4][8][32][16];
  __shared__ float w3t[4 * 128];                      // tail rows 128..131 of w3 (conditions)
  __shared__ float b3s[128], g3s[128], be3s[128];
  __shared__ __align__(16) _Float16 h3A[128 * 136];
  __shared__ __align__(32) _Float16 W4p[4][4][32][16];
  __shared__ float b4s[64], g4s[64], be4s[64], w5s[64];
  __shared__ float conds[128][4];
  __shared__ float red[8];
  float* h4 = (float*)gfA;                            // [128][65] f32 alias (after layer1)

  const int t = threadIdx.x, lane = t & 31, wv = t >> 5;
  const int rowbase = blockIdx.x * 128;

  // ---- stage weights ----
  for (int e = t; e < 4 * 8 * 32 * 16; e += 256) {
    int i = e & 15, ln = (e >> 4) & 31, ct = (e >> 9) & 7, ks = (e >> 12) & 3;
    int k = ks * 32 + ((ln >> 4) << 3) + (i < 8 ? i : 16 + (i - 8));
    int c = ct * 16 + (ln & 15);
    W3p[ks][ct][ln][i] = (_Float16)w3[k * 128 + c];
  }
  for (int e = t; e < 4 * 4 * 32 * 16; e += 256) {
    int i = e & 15, ln = (e >> 4) & 31, ct = (e >> 9) & 3, ks = (e >> 11) & 3;
    int k = ks * 32 + ((ln >> 4) << 3) + (i < 8 ? i : 16 + (i - 8));
    int c = ct * 16 + (ln & 15);
    W4p[ks][ct][ln][i] = (_Float16)w4[k * 64 + c];
  }
  for (int i = t; i < 4 * 128; i += 256) w3t[i] = w3[(128 + (i >> 7)) * 128 + (i & 127)];
  for (int i = t; i < 128; i += 256) { b3s[i] = b3[i]; g3s[i] = g3[i]; be3s[i] = be3[i]; }
  for (int i = t; i < 64; i += 256) { b4s[i] = b4[i]; g4s[i] = g4[i]; be4s[i] = be4[i]; w5s[i] = w5[i]; }

  // ---- decode gf (with nan_to_num fills) + conditions ----
  for (int e = t; e < 128 * 128; e += 256) {
    int r = e >> 7, c = e & 127;
    int gr = rowbase + r; if (gr >= B) gr = B - 1;
    float v = fdec(gfkey[(size_t)gr * 128 + c]);
    if (__builtin_isnan(v)) v = NEG_FILL;
    else if (__builtin_isinf(v)) v = (v > 0.f) ? 1e9f : NEG_FILL;
    gfA[r * 136 + c] = (_Float16)v;
  }
  for (int e = t; e < 128 * 4; e += 256) {
    int r = e >> 2, i = e & 3;
    int gr = rowbase + r; if (gr >= B) gr = B - 1;
    conds[r][i] = cond[(size_t)gr * 4 + i];
  }
  __syncthreads();

  // ---- layer 1: [128 rows] x (gf @ w3[0:128] + cond @ w3[128:132] + b3), LN, lrelu ----
  {
    const int rw  = wv * 16 + (lane & 15);
    const int kb8 = (lane >> 4) << 3;
    v16h A[4];
    #pragma unroll
    for (int ks = 0; ks < 4; ++ks) {
      int kb = ks * 32 + kb8;
      union { v16h v; v8h h[2]; } u;
      u.h[0] = *(const v8h*)&gfA[rw * 136 + kb];
      u.h[1] = *(const v8h*)&gfA[rw * 136 + kb + 16];
      A[ks] = u.v;
    }
    v8f acc[8];
    #pragma unroll
    for (int ct = 0; ct < 8; ++ct) {
      int c = ct * 16 + (lane & 15);
      float b0 = b3s[c];
      float t0 = w3t[0 * 128 + c], t1 = w3t[1 * 128 + c];
      float t2 = w3t[2 * 128 + c], t3 = w3t[3 * 128 + c];
      #pragma unroll
      for (int v = 0; v < 8; ++v) {
        int r = wv * 16 + v + kb8;
        acc[ct][v] = b0 + conds[r][0] * t0 + conds[r][1] * t1
                        + conds[r][2] * t2 + conds[r][3] * t3;
      }
    }
    #pragma unroll
    for (int ct = 0; ct < 8; ++ct) {
      #pragma unroll
      for (int ks = 0; ks < 4; ++ks) {
        union { v16h v; v8h h[2]; } u;
        u.h[0] = *(const v8h*)&W3p[ks][ct][lane][0];
        u.h[1] = *(const v8h*)&W3p[ks][ct][lane][8];
        acc[ct] = __builtin_amdgcn_wmma_f32_16x16x32_f16(
            false, A[ks], false, u.v, (short)0, acc[ct], false, false);
      }
    }
    #pragma unroll
    for (int v = 0; v < 8; ++v) {
      float s = 0.f, sq = 0.f;
      #pragma unroll
      for (int ct = 0; ct < 8; ++ct) { float x = acc[ct][v]; s += x; sq += x * x; }
      s = hsum16(s); sq = hsum16(sq);
      float mm = s * (1.f / 128.f), vv = sq * (1.f / 128.f) - mm * mm;
      float inv = rsqrtf(vv + LN_EPS);
      #pragma unroll
      for (int ct = 0; ct < 8; ++ct) {
        int c = ct * 16 + (lane & 15);
        int r = wv * 16 + v + kb8;
        float x = (acc[ct][v] - mm) * inv * g3s[c] + be3s[c];
        h3A[r * 136 + c] = (_Float16)lrelu(x);
      }
    }
  }
  __syncthreads();

  // ---- layer 2: 128 -> 64, LN, lrelu ----
  {
    const int rw  = wv * 16 + (lane & 15);
    const int kb8 = (lane >> 4) << 3;
    v16h A[4];
    #pragma unroll
    for (int ks = 0; ks < 4; ++ks) {
      int kb = ks * 32 + kb8;
      union { v16h v; v8h h[2]; } u;
      u.h[0] = *(const v8h*)&h3A[rw * 136 + kb];
      u.h[1] = *(const v8h*)&h3A[rw * 136 + kb + 16];
      A[ks] = u.v;
    }
    v8f acc[4];
    #pragma unroll
    for (int ct = 0; ct < 4; ++ct) {
      float bz = b4s[ct * 16 + (lane & 15)];
      v8f a = {bz, bz, bz, bz, bz, bz, bz, bz};
      acc[ct] = a;
    }
    #pragma unroll
    for (int ct = 0; ct < 4; ++ct) {
      #pragma unroll
      for (int ks = 0; ks < 4; ++ks) {
        union { v16h v; v8h h[2]; } u;
        u.h[0] = *(const v8h*)&W4p[ks][ct][lane][0];
        u.h[1] = *(const v8h*)&W4p[ks][ct][lane][8];
        acc[ct] = __builtin_amdgcn_wmma_f32_16x16x32_f16(
            false, A[ks], false, u.v, (short)0, acc[ct], false, false);
      }
    }
    #pragma unroll
    for (int v = 0; v < 8; ++v) {
      float s = 0.f, sq = 0.f;
      #pragma unroll
      for (int ct = 0; ct < 4; ++ct) { float x = acc[ct][v]; s += x; sq += x * x; }
      s = hsum16(s); sq = hsum16(sq);
      float mm = s * (1.f / 64.f), vv = sq * (1.f / 64.f) - mm * mm;
      float inv = rsqrtf(vv + LN_EPS);
      #pragma unroll
      for (int ct = 0; ct < 4; ++ct) {
        int c = ct * 16 + (lane & 15);
        int r = wv * 16 + v + kb8;
        float x = (acc[ct][v] - mm) * inv * g4s[c] + be4s[c];
        h4[r * 65 + c] = lrelu(x);          // gfA region is dead; alias as f32
      }
    }
  }
  __syncthreads();

  // ---- layer 3: 64 -> 1, then block-reduce |score| ----
  float myabs = 0.f;
  if (t < 128) {
    float s = b5[0];
    #pragma unroll 8
    for (int i = 0; i < 64; ++i) s += h4[t * 65 + i] * w5s[i];
    int gr = rowbase + t;
    if (gr < B) score[gr] = s;
    myabs = fabsf(s);
  }
  myabs += __shfl_xor(myabs, 1);
  myabs += __shfl_xor(myabs, 2);
  myabs += __shfl_xor(myabs, 4);
  myabs += __shfl_xor(myabs, 8);
  myabs += __shfl_xor(myabs, 16);
  if (lane == 0) red[wv] = myabs;
  __syncthreads();
  if (t == 0) {
    float s = 0.f;
    #pragma unroll
    for (int i = 0; i < 8; ++i) s += red[i];
    atomicAdd(sumAbs, s);
  }
}

// ---------------------------------------------------------------------------
// Kernel 3: EMA normalization + tanh squash
// ---------------------------------------------------------------------------
__global__ void k_final(const float* __restrict__ score, const float* __restrict__ sumAbs,
                        float* __restrict__ out, int B) {
  int i = blockIdx.x * blockDim.x + threadIdx.x;
  if (i >= B) return;
  float ema   = 0.99f + 0.01f * (sumAbs[0] / (float)B);
  float denom = fmaxf(ema, 0.1f);
  float s = score[i] / denom;
  s = fminf(fmaxf(s, -5.f), 5.f);
  out[i] = 10.f * tanhf(s * 0.2f);
}

// ---------------------------------------------------------------------------
extern "C" void kernel_launch(void* const* d_in, const int* in_sizes, int n_in,
                              void* d_out, int out_size, void* d_ws, size_t ws_size,
                              hipStream_t stream) {
  const float* muons  = (const float*)d_in[0];
  const int*   bindex = (const int*)d_in[1];
  const float* cond   = (const float*)d_in[2];
  // d_in[3] = batch_size scalar (derived from in_sizes instead)
  const float* w1  = (const float*)d_in[4];
  const float* b1  = (const float*)d_in[5];
  const float* g1  = (const float*)d_in[6];
  const float* be1 = (const float*)d_in[7];
  const float* w2  = (const float*)d_in[8];
  const float* b2  = (const float*)d_in[9];
  const float* g2  = (const float*)d_in[10];
  const float* be2 = (const float*)d_in[11];
  const float* w3  = (const float*)d_in[12];
  const float* b3  = (const float*)d_in[13];
  const float* g3  = (const float*)d_in[14];
  const float* be3 = (const float*)d_in[15];
  const float* w4  = (const float*)d_in[16];
  const float* b4  = (const float*)d_in[17];
  const float* g4  = (const float*)d_in[18];
  const float* be4 = (const float*)d_in[19];
  const float* w5  = (const float*)d_in[20];
  const float* b5  = (const float*)d_in[21];

  const int N = in_sizes[0] / 3;
  const int B = in_sizes[2] / 4;

  unsigned* gfkey  = (unsigned*)d_ws;
  float*    score  = (float*)((char*)d_ws + (size_t)B * 128 * sizeof(unsigned));
  float*    sumAbs = (float*)((char*)d_ws + (size_t)B * 128 * sizeof(unsigned)
                                          + (size_t)B * sizeof(float));

  const int tot = B * 128;
  k_init<<<(tot + 255) / 256, 256, 0, stream>>>(gfkey, sumAbs, tot);
  k_pointnet<<<(N + 127) / 128, 256, 0, stream>>>(muons, bindex, cond,
                                                  w1, b1, g1, be1, w2, b2, g2, be2,
                                                  gfkey, N);
  k_decision<<<(B + 127) / 128, 256, 0, stream>>>(gfkey, cond,
                                                  w3, b3, g3, be3, w4, b4, g4, be4,
                                                  w5, b5, score, sumAbs, B);
  k_final<<<(B + 255) / 256, 256, 0, stream>>>(score, sumAbs, (float*)d_out, B);
}
